// DownBlock_40802189312117
// MI455X (gfx1250) — compile-verified, matrix operand order checked by reading, not run on hardware
//
#include <hip/hip_runtime.h>
#include <hip/hip_bf16.h>
#include <math.h>

#define B_    2
#define CIN   64
#define COUT  128
#define T_    8192
#define LSEQ  4096
#define KCONV 5
#define NM    2
#define NST   16
#define DCONV 4
#define DI    256
#define DTR   8
#define NXP   64            // padded DTR + 2*NST = 40 -> 64 (all GEMM N % 64 == 0)
#define BL    (B_*LSEQ)     // 8192 rows
#define GEMM_MAXK 256

typedef __attribute__((ext_vector_type(16))) __bf16 v16bf;
typedef __attribute__((ext_vector_type(8)))  float  v8f;
typedef __attribute__((ext_vector_type(4)))  unsigned v4u;
typedef __attribute__((ext_vector_type(8)))  int      v8i;
typedef __attribute__((ext_vector_type(4)))  int      v4i;

union BfBits { unsigned short u; __bf16 b; };
__device__ inline __bf16 bf_from_bits(unsigned short s){ BfBits c; c.u=s; return c.b; }
__device__ inline unsigned short f2bf_bits(float f){
  union { float f; unsigned u; } v; v.f=f;
  unsigned r = v.u + 0x7fffu + ((v.u>>16)&1u);
  return (unsigned short)(r>>16);
}
__device__ inline float sigmoidf_(float x){ return 1.0f/(1.0f+__expf(-x)); }

// ---------------- front conv (stride 2) + GN partial sums -------------------
__global__ void k_front_conv(const float* __restrict__ x, const float* __restrict__ w,
                             const float* __restrict__ bias, float* __restrict__ h,
                             float* __restrict__ part){
  int b = blockIdx.y;
  int idx = blockIdx.x*256 + threadIdx.x;          // over LSEQ*COUT
  float val = 0.f;
  if (idx < LSEQ*COUT) {
    int l = idx / COUT, c = idx % COUT;
    float acc = bias[c];
    const float* xb = x + (size_t)b*CIN*T_;
    const float* wc = w + (size_t)c*CIN*KCONV;
    int t0 = 2*l - (KCONV/2);
    #pragma unroll
    for (int k=0;k<KCONV;k++){
      int t = t0 + k;
      if (t>=0 && t<T_){
        for (int ci=0; ci<CIN; ci++)
          acc += xb[(size_t)ci*T_ + t] * wc[ci*KCONV + k];
      }
    }
    h[((size_t)b*LSEQ + l)*COUT + c] = acc;
    val = acc;
  }
  __shared__ float s1[256], s2[256];
  s1[threadIdx.x] = (idx < LSEQ*COUT) ? val : 0.f;
  s2[threadIdx.x] = (idx < LSEQ*COUT) ? val*val : 0.f;
  __syncthreads();
  for (int s=128;s>0;s>>=1){
    if (threadIdx.x<s){ s1[threadIdx.x]+=s1[threadIdx.x+s]; s2[threadIdx.x]+=s2[threadIdx.x+s]; }
    __syncthreads();
  }
  if (threadIdx.x==0){
    part[((size_t)b*gridDim.x + blockIdx.x)*2+0] = s1[0];
    part[((size_t)b*gridDim.x + blockIdx.x)*2+1] = s2[0];
  }
}

__global__ void k_gn_finalize(const float* __restrict__ part, float* __restrict__ stats, int nPart){
  int b = blockIdx.x, t = threadIdx.x;
  __shared__ float s1[256], s2[256];
  float a=0.f, q=0.f;
  for (int i=t;i<nPart;i+=256){ a+=part[((size_t)b*nPart+i)*2]; q+=part[((size_t)b*nPart+i)*2+1]; }
  s1[t]=a; s2[t]=q; __syncthreads();
  for (int s=128;s>0;s>>=1){ if(t<s){s1[t]+=s1[t+s]; s2[t]+=s2[t+s];} __syncthreads(); }
  if (t==0){
    float inv = 1.0f/((float)LSEQ*COUT);
    float mu = s1[0]*inv;
    float var = s2[0]*inv - mu*mu;
    stats[b*2] = mu; stats[b*2+1] = rsqrtf(var + 1e-5f);
  }
}

__global__ void k_gn_prelu(float* __restrict__ h, const float* __restrict__ stats,
                           const float* __restrict__ g, const float* __restrict__ bb,
                           const float* __restrict__ pa){
  int idx = blockIdx.x*256 + threadIdx.x;
  if (idx >= BL*COUT) return;
  int c = idx % COUT;
  int b = idx / (LSEQ*COUT);
  float v = (h[idx] - stats[b*2]) * stats[b*2+1] * g[c] + bb[c];
  h[idx] = (v >= 0.f) ? v : pa[0]*v;
}

// ---------------- one-shot weight conversion to bf16 ------------------------
__global__ void k_conv_weights(const float* __restrict__ Win, const float* __restrict__ Wout,
                               const float* __restrict__ Wx,
                               unsigned short* __restrict__ WinB, unsigned short* __restrict__ WoutB,
                               unsigned short* __restrict__ WxB){
  int idx = blockIdx.x*256 + threadIdx.x;
  const int nWin  = 4*2*DI*COUT;   // 262144
  const int nWout = 4*COUT*DI;     // 131072
  const int nWx   = 4*NXP*DI;      // 65536 (zero-padded rows 40..63)
  if (idx < nWin){ WinB[idx] = f2bf_bits(Win[idx]); return; }
  idx -= nWin;
  if (idx < nWout){ WoutB[idx] = f2bf_bits(Wout[idx]); return; }
  idx -= nWout;
  if (idx < nWx){
    int m = idx / (NXP*DI), r = idx % (NXP*DI), e = r / DI, d = r % DI;
    float v = (e < DTR + 2*NST) ? Wx[((size_t)m*(DTR+2*NST)+e)*DI + d] : 0.f;
    WxB[idx] = f2bf_bits(v);
  }
}

// ---------------- per-row layernorm -> bf16 ---------------------------------
__global__ void k_layernorm(const float* __restrict__ h, const float* __restrict__ g,
                            const float* __restrict__ bb, unsigned short* __restrict__ hnB){
  int row = blockIdx.x*8 + (threadIdx.x>>5);
  int lane = threadIdx.x & 31;
  if (row >= BL) return;
  const float* hr = h + (size_t)row*COUT;
  float v[4]; float s = 0.f;
  #pragma unroll
  for (int j=0;j<4;j++){ v[j] = hr[lane + 32*j]; s += v[j]; }
  #pragma unroll
  for (int m=16;m>0;m>>=1) s += __shfl_xor(s, m, 32);
  float mu = s * (1.0f/COUT);
  float q = 0.f;
  #pragma unroll
  for (int j=0;j<4;j++){ float d = v[j]-mu; q += d*d; }
  #pragma unroll
  for (int m=16;m>0;m>>=1) q += __shfl_xor(q, m, 32);
  float rs = rsqrtf(q*(1.0f/COUT) + 1e-5f);
  #pragma unroll
  for (int j=0;j<4;j++){
    int c = lane + 32*j;
    hnB[(size_t)row*COUT + c] = f2bf_bits((v[j]-mu)*rs*g[c] + bb[c]);
  }
}

// ---------------- WMMA bf16 GEMM: Out[M,N] (+)= A[M,K] * W[N,K]^T -----------
// Block = one 16-row strip of A (TDM-staged into LDS once), waves = Ncols/64,
// each wave register-blocks 4 N-tiles (16x64) so one A fragment feeds 4 WMMAs.
// flipRead/flipWrite fold the sequence reversal of the backward mambas in.
__global__ void __launch_bounds__(256) k_wmma_gemm(
    const unsigned short* __restrict__ A, const unsigned short* __restrict__ W,
    float* __restrict__ Out, int Kdim, int Ncols,
    int flipRead, int flipWrite, int accumulate)
{
  __shared__ unsigned short lds_a[16*GEMM_MAXK];
  int tm   = blockIdx.x;
  int wave = threadIdx.x >> 5;
  int lane = threadIdx.x & 31;

  // --- stage A strip (16 x Kdim) into LDS (contiguous even when flipped) ----
  int row0 = tm*16;
  if (flipRead){ int b = row0 >> 12; int l = row0 & (LSEQ-1); row0 = (b<<12) + (LSEQ-16 - l); }
  const unsigned short* atile = A + (size_t)row0*Kdim;

#if defined(__has_builtin) && __has_builtin(__builtin_amdgcn_tensor_load_to_lds)
  if (threadIdx.x < 32){                       // one wave drives the TDM
    unsigned long long ga = (unsigned long long)(size_t)atile;
    unsigned ldsAddr = (unsigned)(size_t)(void*)lds_a;   // LDS aperture low bits
    // D# group0: count=1 | lds_addr | global_addr[56:0] | type=2
    v4u g0 = { 1u, ldsAddr, (unsigned)ga,
               (unsigned)((ga >> 32) & 0x01ffffffu) | 0x80000000u };
    // D# group1: data_size=2B; tensor_dim0=Kdim; tensor_dim1=16;
    // tile_dim0=Kdim; tile_dim1=16; dim0_stride=Kdim
    v8i g1 = { (int)(1u << 16),              // data_size=1 (2 bytes)
               (int)((unsigned)Kdim << 16),  // tensor_dim0[15:0]
               (int)(16u << 16),             // tensor_dim1[15:0]
               (int)((unsigned)Kdim << 16),  // tile_dim0
               16,                           // tile_dim1 (tile_dim2=0)
               Kdim,                         // tensor_dim0_stride[31:0]
               0, 0 };
    v4i g2 = { 0, 0, 0, 0 };
    v4i g3 = { 0, 0, 0, 0 };
#if __clang_major__ >= 23
    v8i g4 = { 0, 0, 0, 0, 0, 0, 0, 0 };
    __builtin_amdgcn_tensor_load_to_lds(g0, g1, g2, g3, g4, 0);
#else
    __builtin_amdgcn_tensor_load_to_lds(g0, g1, g2, g3, 0);
#endif
    __builtin_amdgcn_s_wait_tensorcnt(0);
  }
#else
  for (int i = threadIdx.x; i < (16*Kdim)/2; i += blockDim.x)
    ((unsigned*)lds_a)[i] = ((const unsigned*)atile)[i];
#endif
  __syncthreads();

  // --- compute: each wave owns cols [wave*64, wave*64+64) -------------------
  int mrow = lane & 15;
  int kgrp = (lane >> 4) & 1;
  int arow_l = flipRead ? (15 - mrow) : mrow;    // un-reverse flipped strip
  const unsigned short* arow = lds_a + (size_t)arow_l * Kdim;

  int ncol0 = wave*64;
  const unsigned short* wrow = W + (size_t)(ncol0 + (lane & 15)) * Kdim;

  v8f acc0 = {}, acc1 = {}, acc2 = {}, acc3 = {};
  for (int k0 = 0; k0 < Kdim; k0 += 32){
    v16bf a, b0, b1, b2, b3;
    #pragma unroll
    for (int i=0;i<8;i++){
      int k = k0 + ((i>>2)<<4) + (kgrp<<3) + ((i&3)<<1);  // 16-bit A/B striping
      unsigned av  = *(const unsigned*)(arow + k);
      unsigned bv0 = *(const unsigned*)(wrow + k);
      unsigned bv1 = *(const unsigned*)(wrow + 16*Kdim + k);
      unsigned bv2 = *(const unsigned*)(wrow + 32*Kdim + k);
      unsigned bv3 = *(const unsigned*)(wrow + 48*Kdim + k);
      a [2*i] = bf_from_bits((unsigned short)(av  & 0xffffu)); a [2*i+1] = bf_from_bits((unsigned short)(av  >> 16));
      b0[2*i] = bf_from_bits((unsigned short)(bv0 & 0xffffu)); b0[2*i+1] = bf_from_bits((unsigned short)(bv0 >> 16));
      b1[2*i] = bf_from_bits((unsigned short)(bv1 & 0xffffu)); b1[2*i+1] = bf_from_bits((unsigned short)(bv1 >> 16));
      b2[2*i] = bf_from_bits((unsigned short)(bv2 & 0xffffu)); b2[2*i+1] = bf_from_bits((unsigned short)(bv2 >> 16));
      b3[2*i] = bf_from_bits((unsigned short)(bv3 & 0xffffu)); b3[2*i+1] = bf_from_bits((unsigned short)(bv3 >> 16));
    }
    acc0 = __builtin_amdgcn_wmma_f32_16x16x32_bf16(false, a, false, b0, (short)0, acc0, false, false);
    acc1 = __builtin_amdgcn_wmma_f32_16x16x32_bf16(false, a, false, b1, (short)0, acc1, false, false);
    acc2 = __builtin_amdgcn_wmma_f32_16x16x32_bf16(false, a, false, b2, (short)0, acc2, false, false);
    acc3 = __builtin_amdgcn_wmma_f32_16x16x32_bf16(false, a, false, b3, (short)0, acc3, false, false);
  }

  int n = lane & 15, hi = lane >> 4;
  #pragma unroll
  for (int r=0;r<8;r++){
    int row = tm*16 + r + 8*hi;                  // C/D layout: VGPR r -> M=r / M=8+r
    if (flipWrite){ int b = row >> 12; int l = row & (LSEQ-1); row = (b<<12) | (LSEQ-1-l); }
    float* o = Out + (size_t)row*Ncols + ncol0 + n;
    if (accumulate){ o[0] += acc0[r]; o[16] += acc1[r]; o[32] += acc2[r]; o[48] += acc3[r]; }
    else           { o[0]  = acc0[r]; o[16]  = acc1[r]; o[32]  = acc2[r]; o[48]  = acc3[r]; }
  }
}

// ---------------- depthwise causal conv (DCONV=4) + SiLU --------------------
__global__ void k_dwconv_silu(const float* __restrict__ xz, const float* __restrict__ cw,
                              const float* __restrict__ cb, float* __restrict__ uf,
                              unsigned short* __restrict__ ub, int m){
  int idx = blockIdx.x*256 + threadIdx.x;         // BL*DI
  if (idx >= BL*DI) return;
  int d = idx % DI, r = idx / DI, l = r & (LSEQ-1);
  const float* w = cw + ((size_t)m*DI + d)*DCONV;
  float acc = cb[m*DI + d];
  #pragma unroll
  for (int j=0;j<DCONV;j++){
    int ls = l - (DCONV-1) + j;
    if (ls >= 0) acc += xz[((size_t)(r - l + ls))*(2*DI) + d] * w[j];
  }
  float u = acc * sigmoidf_(acc);
  uf[idx] = u;
  ub[idx] = f2bf_bits(u);
}

// ---------------- delta = softplus(dt @ W_dt^T + b_dt) ----------------------
__global__ void k_delta(const float* __restrict__ xdbl, const float* __restrict__ Wd,
                        const float* __restrict__ bd, float* __restrict__ delta, int m){
  int idx = blockIdx.x*256 + threadIdx.x;
  if (idx >= BL*DI) return;
  int d = idx % DI, r = idx / DI;
  const float* dtp = xdbl + (size_t)r*NXP;
  const float* wd  = Wd + ((size_t)m*DI + d)*DTR;
  float acc = bd[m*DI + d];
  #pragma unroll
  for (int j=0;j<DTR;j++) acc += dtp[j]*wd[j];
  delta[idx] = (acc > 20.f) ? acc : log1pf(__expf(acc));
}

// ---------------- selective scan (fused D-skip + SiLU(z) gate) --------------
// 2 channels per wave: lanes 0-15 -> state dims of channel ch, 16-31 -> ch+1.
__global__ void __launch_bounds__(256) k_scan(
    const float* __restrict__ delta, const float* __restrict__ uf,
    const float* __restrict__ xdbl, const float* __restrict__ xz,
    const float* __restrict__ Alog, const float* __restrict__ Dpar,
    unsigned short* __restrict__ yb, int m)
{
  int w = blockIdx.x*8 + (threadIdx.x>>5);
  int lane = threadIdx.x & 31;
  int ch = w*2 + (lane>>4);
  if (ch >= B_*DI) return;
  int b = ch / DI, d = ch % DI, n = lane & 15;
  float Aval = -__expf(Alog[((size_t)m*DI + d)*NST + n]);
  float Dp   = Dpar[m*DI + d];
  float hst = 0.f;
  size_t base = (size_t)b*LSEQ;
  for (int l=0; l<LSEQ; l++){
    size_t r = base + l;
    if (l + 8 < LSEQ){                                    // global_prefetch_b8 ahead
      __builtin_prefetch(&delta[(r+8)*DI + d], 0, 0);
      __builtin_prefetch(&xdbl [(r+8)*NXP + DTR + n], 0, 0);
    }
    float dv = delta[r*DI + d];
    float uv = uf   [r*DI + d];
    float Bv = xdbl [r*NXP + DTR + n];
    float Cv = xdbl [r*NXP + DTR + NST + n];
    float dA = __expf(dv * Aval);
    hst = dA*hst + (dv*uv)*Bv;
    float p = hst * Cv;                                   // reduce over 16 state lanes
    p += __shfl_xor(p, 8, 32);
    p += __shfl_xor(p, 4, 32);
    p += __shfl_xor(p, 2, 32);
    p += __shfl_xor(p, 1, 32);
    if (n == 0){
      float z  = xz[r*(2*DI) + DI + d];
      float yv = (p + uv*Dp) * (z * sigmoidf_(z));
      yb[r*DI + d] = f2bf_bits(yv);
    }
  }
}

// ---------------- [B,L,C] -> [B,C,L] ----------------------------------------
__global__ void k_transpose_out(const float* __restrict__ h, float* __restrict__ out){
  int idx = blockIdx.x*256 + threadIdx.x;
  if (idx >= BL*COUT) return;
  int c = idx % COUT, r = idx / COUT, b = r >> 12, l = r & (LSEQ-1);
  out[((size_t)b*COUT + c)*LSEQ + l] = h[idx];
}

extern "C" void kernel_launch(void* const* d_in, const int* in_sizes, int n_in,
                              void* d_out, int out_size, void* d_ws, size_t ws_size,
                              hipStream_t stream){
  const float* x      = (const float*)d_in[0];
  const float* conv_w = (const float*)d_in[1];
  const float* conv_b = (const float*)d_in[2];
  const float* gn_g   = (const float*)d_in[3];
  const float* gn_b   = (const float*)d_in[4];
  const float* prelu  = (const float*)d_in[5];
  const float* ln_g   = (const float*)d_in[6];
  const float* ln_b   = (const float*)d_in[7];
  const float* W_in   = (const float*)d_in[8];
  const float* c1w    = (const float*)d_in[9];
  const float* c1b    = (const float*)d_in[10];
  const float* W_x    = (const float*)d_in[11];
  const float* W_dt   = (const float*)d_in[12];
  const float* b_dt   = (const float*)d_in[13];
  const float* A_log  = (const float*)d_in[14];
  const float* D_par  = (const float*)d_in[15];
  const float* W_out  = (const float*)d_in[16];
  float* out = (float*)d_out;

  char* ws = (char*)d_ws;
  size_t off = 0;
  auto take = [&](size_t bytes)->char*{
    char* p = ws + off; off = (off + bytes + 255) & ~(size_t)255; return p;
  };
  float*          h    = (float*)         take(sizeof(float)*BL*COUT);
  unsigned short* hnB  = (unsigned short*)take(sizeof(short)*BL*COUT);
  float*          xz   = (float*)         take(sizeof(float)*(size_t)BL*2*DI);
  float*          uf   = (float*)         take(sizeof(float)*(size_t)BL*DI);
  unsigned short* ub   = (unsigned short*)take(sizeof(short)*(size_t)BL*DI);
  float*          xdbl = (float*)         take(sizeof(float)*(size_t)BL*NXP);
  float*          delta= (float*)         take(sizeof(float)*(size_t)BL*DI);
  unsigned short* ybB  = (unsigned short*)take(sizeof(short)*(size_t)BL*DI);
  unsigned short* WinB = (unsigned short*)take(sizeof(short)*4*2*DI*COUT);
  unsigned short* WoutB= (unsigned short*)take(sizeof(short)*4*COUT*DI);
  unsigned short* WxB  = (unsigned short*)take(sizeof(short)*4*NXP*DI);
  float*          part = (float*)         take(sizeof(float)*B_*2048*2);
  float*          stats= (float*)         take(sizeof(float)*8);

  int nConv = 4*2*DI*COUT + 4*COUT*DI + 4*NXP*DI;
  k_conv_weights<<<(nConv+255)/256, 256, 0, stream>>>(W_in, W_out, W_x, WinB, WoutB, WxB);

  dim3 gc(2048, B_);   // (LSEQ*COUT)/256 == 2048 exactly
  k_front_conv<<<gc, 256, 0, stream>>>(x, conv_w, conv_b, h, part);
  k_gn_finalize<<<B_, 256, 0, stream>>>(part, stats, 2048);
  k_gn_prelu<<<(BL*COUT+255)/256, 256, 0, stream>>>(h, stats, gn_g, gn_b, prelu);

  for (int i=0;i<NM;i++){
    k_layernorm<<<BL/8, 256, 0, stream>>>(h, ln_g + i*COUT, ln_b + i*COUT, hnB);
    for (int dir=0; dir<2; dir++){
      int m = 2*i + dir;
      // in-proj: hn[8192,128] x W_in[512,128]^T -> xz[8192,512]
      k_wmma_gemm<<<BL/16, 32*(2*DI/64), 0, stream>>>(hnB, WinB + (size_t)m*2*DI*COUT, xz,
          COUT, 2*DI, dir, 0, 0);
      k_dwconv_silu<<<(BL*DI+255)/256, 256, 0, stream>>>(xz, c1w, c1b, uf, ub, m);
      // x-proj: u[8192,256] x Wx[64,256]^T -> xdbl[8192,64]
      k_wmma_gemm<<<BL/16, 32*(NXP/64), 0, stream>>>(ub, WxB + (size_t)m*NXP*DI, xdbl,
          DI, NXP, 0, 0, 0);
      k_delta<<<(BL*DI+255)/256, 256, 0, stream>>>(xdbl, W_dt, b_dt, delta, m);
      k_scan<<<(B_*DI/2)/8, 256, 0, stream>>>(delta, uf, xdbl, xz, A_log, D_par, ybB, m);
      // out-proj: y[8192,256] x W_out[128,256]^T accumulated into h (flip write for bwd)
      k_wmma_gemm<<<BL/16, 32*(COUT/64), 0, stream>>>(ybB, WoutB + (size_t)m*COUT*DI, h,
          DI, COUT, 0, dir, 1);
    }
  }
  k_transpose_out<<<(BL*COUT+255)/256, 256, 0, stream>>>(h, out);
}
